// CrossCoder_3831110828647
// MI455X (gfx1250) — matrix operand order repaired and checked by reading, not compiled, not run on hardware
//
#include <hip/hip_runtime.h>
#include <hip/hip_bf16.h>

// Problem sizes (fixed by the reference)
#define B_ROWS 4096
#define D_DIM  4096
#define H_DIM  16384
#define K_TOP  64

// GEMM tile sizes: 256(M) x 64(N) workgroup tile, K chunk 64.
// (B/BM)=16 passes over the 256MB weight matrix -> ~4 GB HBM weight traffic.
#define BM 256
#define BN 64
#define BK 64

typedef __attribute__((ext_vector_type(16))) __bf16 v16bf;
typedef __attribute__((ext_vector_type(8)))  __bf16 bf16x8;
typedef __attribute__((ext_vector_type(8)))  float  v8f;

// ---------------------------------------------------------------------------
// Kernel 1: pre = x @ dec^T + enc_bias   via v_wmma_f32_16x16x32_bf16
//   (reference guarantees enc == dec.T, and dec[H,D] row-major is exactly the
//    [n][k] layout the CDNA5 B-fragment wants -> transpose-free staging)
//   - 256 threads = 8 wave32s; each wave owns 32(M) x 64(N) = 2x4 C tiles
//   - both tiles staged as bf16 in LDS with contiguous-K rows, so A and B
//     fragments are contiguous 16B LDS reads per ISA 7.12.2 layouts.
// ---------------------------------------------------------------------------
__global__ __launch_bounds__(256) void cc_encode_wmma(
    const float* __restrict__ x, const float* __restrict__ dec,
    const float* __restrict__ enc_bias, float* __restrict__ pre)
{
    __shared__ alignas(32) __bf16 xA[BM * BK];   // 32 KB  [m][k]
    __shared__ alignas(32) __bf16 dB[BN * BK];   //  8 KB  [n][k]

    const int t     = threadIdx.x;
    const int w     = t >> 5;          // wave 0..7
    const int lane  = t & 31;
    const int l15   = lane & 15;
    const int lhalf = lane >> 4;       // 0 or 1
    const int m0    = blockIdx.y * BM;
    const int n0    = blockIdx.x * BN;

    v8f acc[2][4];
#pragma unroll
    for (int mt = 0; mt < 2; ++mt)
#pragma unroll
        for (int nt = 0; nt < 4; ++nt)
            acc[mt][nt] = (v8f){0.f,0.f,0.f,0.f,0.f,0.f,0.f,0.f};

    for (int k0 = 0; k0 < D_DIM; k0 += BK) {
        // --- stage x tile [BM x BK] f32->bf16, coalesced float4 reads ------
#pragma unroll
        for (int i = 0; i < 16; ++i) {
            const int f  = t + i * 256;     // 0..4095 float4 slots
            const int r  = f >> 4;          // row   0..255
            const int c4 = f & 15;          // col/4 0..15
            const float4 v = *(const float4*)(x + (size_t)(m0 + r) * D_DIM + k0 + c4 * 4);
            __bf16* dst = &xA[r * BK + c4 * 4];
            dst[0] = (__bf16)v.x; dst[1] = (__bf16)v.y;
            dst[2] = (__bf16)v.z; dst[3] = (__bf16)v.w;
        }
        // --- stage dec tile [BN x BK] f32->bf16 (already [n][k] major) -----
#pragma unroll
        for (int i = 0; i < 4; ++i) {
            const int f  = t + i * 256;     // 0..1023 float4 slots
            const int r  = f >> 4;          // n row 0..63
            const int c4 = f & 15;
            const float4 v = *(const float4*)(dec + (size_t)(n0 + r) * D_DIM + k0 + c4 * 4);
            __bf16* dst = &dB[r * BK + c4 * 4];
            dst[0] = (__bf16)v.x; dst[1] = (__bf16)v.y;
            dst[2] = (__bf16)v.z; dst[3] = (__bf16)v.w;
        }
        // prefetch next K-slice of the weight tile into L2 (global_prefetch_b8)
        if (k0 + BK < D_DIM)
            __builtin_prefetch(dec + (size_t)(n0 + (t >> 2)) * D_DIM + k0 + BK + (t & 3) * 16, 0, 1);
        __syncthreads();

        // --- WMMA: 2 k-steps x (2 M-tiles x 4 N-tiles) per wave ------------
#pragma unroll
        for (int kk = 0; kk < BK; kk += 32) {
            v16bf a[2], bf[4];
#pragma unroll
            for (int mt = 0; mt < 2; ++mt) {
                // A fragment (16x32): lanes 0-15 K=kk+0..7 & kk+16..23,
                //                     lanes 16-31 K=kk+8..15 & kk+24..31
                const __bf16* ap = &xA[(w * 32 + mt * 16 + l15) * BK + kk + lhalf * 8];
                bf16x8 alo = *(const bf16x8*)ap;
                bf16x8 ahi = *(const bf16x8*)(ap + 16);
                a[mt] = __builtin_shufflevector(alo, ahi,
                            0,1,2,3,4,5,6,7,8,9,10,11,12,13,14,15);
            }
#pragma unroll
            for (int nt = 0; nt < 4; ++nt) {
                // B fragment (32x16): lane = column N, lanes 0-15 K=kk+0..7 & kk+16..23,
                //                     lanes 16-31 K=kk+8..15 & kk+24..31
                const __bf16* bp = &dB[(nt * 16 + l15) * BK + kk + lhalf * 8];
                bf16x8 blo = *(const bf16x8*)bp;
                bf16x8 bhi = *(const bf16x8*)(bp + 16);
                bf[nt] = __builtin_shufflevector(blo, bhi,
                            0,1,2,3,4,5,6,7,8,9,10,11,12,13,14,15);
            }
#pragma unroll
            for (int mt = 0; mt < 2; ++mt)
#pragma unroll
                for (int nt = 0; nt < 4; ++nt)
                    acc[mt][nt] = __builtin_amdgcn_wmma_f32_16x16x32_bf16(
                        false, a[mt], false, bf[nt], (short)0, acc[mt][nt], false, false);
        }
        __syncthreads();
    }

    // --- epilogue: fuse bias, scatter per the C/D layout -------------------
    // element r of lane L: M = m0 + w*32 + mt*16 + r + 8*(L>>4),
    //                      N = n0 + nt*16 + (L&15)
#pragma unroll
    for (int nt = 0; nt < 4; ++nt) {
        const int n    = n0 + nt * 16 + l15;
        const float bs = enc_bias[n];
#pragma unroll
        for (int mt = 0; mt < 2; ++mt) {
            const int mb = m0 + w * 32 + mt * 16 + lhalf * 8;
#pragma unroll
            for (int r = 0; r < 8; ++r)
                pre[(size_t)(mb + r) * H_DIM + n] = acc[mt][nt][r] + bs;
        }
    }
}

// ---------------------------------------------------------------------------
// Kernel 2: top-64 per row. One 256-thread WG per row. Row cached in LDS as
// bf16 (32 KB) for the argmax scans; exact f32 value re-read from pre for the
// selected index. ReLU fused into the stored value.
// ---------------------------------------------------------------------------
__global__ __launch_bounds__(256) void cc_topk(
    const float* __restrict__ pre, float* __restrict__ topv, int* __restrict__ topi)
{
    __shared__ __bf16 row[H_DIM];      // 32 KB
    __shared__ float  wmaxv[8];
    __shared__ int    wmaxi[8];

    const int b    = blockIdx.x;
    const int t    = threadIdx.x;
    const int w    = t >> 5;
    const int lane = t & 31;
    const float* prow = pre + (size_t)b * H_DIM;

#pragma unroll
    for (int i = 0; i < 16; ++i) {
        const int f = t + i * 256;                 // float4 slot 0..4095
        const float4 v = *(const float4*)(prow + f * 4);
        row[f * 4 + 0] = (__bf16)v.x; row[f * 4 + 1] = (__bf16)v.y;
        row[f * 4 + 2] = (__bf16)v.z; row[f * 4 + 3] = (__bf16)v.w;
    }
    __syncthreads();

    for (int it = 0; it < K_TOP; ++it) {
        float bv = -3.4e38f; int bi = 0;
#pragma unroll 8
        for (int i = 0; i < H_DIM / 256; ++i) {
            const int idx = t + i * 256;
            const float v = (float)row[idx];
            if (v > bv) { bv = v; bi = idx; }
        }
        // wave32 reduction
#pragma unroll
        for (int off = 16; off > 0; off >>= 1) {
            const float ov = __shfl_xor(bv, off, 32);
            const int   oi = __shfl_xor(bi, off, 32);
            if (ov > bv) { bv = ov; bi = oi; }
        }
        if (lane == 0) { wmaxv[w] = bv; wmaxi[w] = bi; }
        __syncthreads();
        if (t == 0) {
            float gv = wmaxv[0]; int gi = wmaxi[0];
#pragma unroll
            for (int k = 1; k < 8; ++k)
                if (wmaxv[k] > gv) { gv = wmaxv[k]; gi = wmaxi[k]; }
            const float exact = prow[gi];
            topv[b * K_TOP + it] = exact > 0.f ? exact : 0.f;   // fused ReLU
            topi[b * K_TOP + it] = gi;
            row[gi] = (__bf16)(-3.4e38f);   // rounds to -inf bf16: masked
        }
        __syncthreads();
    }
}

// ---------------------------------------------------------------------------
// Kernel 3: sparse decoder with async double-buffered row gather:
//   out[b,:] = sum_j topv[j] * dec[topi[j],:] + dec_bias
// Row j+1 is copied memory->LDS with GLOBAL_LOAD_ASYNC_TO_LDS_B128 (ASYNCcnt)
// while row j is FMA'd out of LDS, hiding the random-gather latency.
// ---------------------------------------------------------------------------
__device__ __forceinline__ void cc_issue_row_async(const float* gsrc, float* lbuf, int t)
{
#pragma unroll
    for (int i = 0; i < 4; ++i) {
        // each thread copies one 16B chunk per issue; 256 thr x 4 x 16B = 16 KB
        const unsigned loff =
            (unsigned)(unsigned long long)(uintptr_t)(lbuf + i * 1024 + t * 4);
        const unsigned long long gaddr =
            (unsigned long long)(uintptr_t)(gsrc + i * 1024 + t * 4);
        asm volatile("global_load_async_to_lds_b128 %0, %1, off"
                     :: "v"(loff), "v"(gaddr) : "memory");
    }
}

__global__ __launch_bounds__(256) void cc_decode_async(
    const float* __restrict__ topv, const int* __restrict__ topi,
    const float* __restrict__ dec, const float* __restrict__ dec_bias,
    float* __restrict__ out)
{
    __shared__ alignas(16) float rowbuf[2][D_DIM];   // 2 x 16 KB double buffer
    __shared__ float sv[K_TOP];
    __shared__ int   si[K_TOP];

    const int b = blockIdx.x;
    const int t = threadIdx.x;
    if (t < K_TOP) { sv[t] = topv[b * K_TOP + t]; si[t] = topi[b * K_TOP + t]; }
    __syncthreads();

    // prologue: async-copy decoder row 0 into buffer 0
    cc_issue_row_async(dec + (size_t)si[0] * D_DIM, rowbuf[0], t);

    float4 acc[4];
#pragma unroll
    for (int s = 0; s < 4; ++s) acc[s] = make_float4(0.f, 0.f, 0.f, 0.f);

    for (int j = 0; j < K_TOP; ++j) {
        // wait for row j's async copies (issued last iteration / prologue)
        asm volatile("s_wait_asynccnt 0x0" ::: "memory");
        __syncthreads();   // all waves' copies of row j landed; row j-1 compute done
        if (j + 1 < K_TOP)
            cc_issue_row_async(dec + (size_t)si[j + 1] * D_DIM,
                               rowbuf[(j + 1) & 1], t);    // overlaps compute below
        const float val = sv[j];       // 0 if relu-killed: harmless multiply
        const float* rb = rowbuf[j & 1];
#pragma unroll
        for (int s = 0; s < 4; ++s) {
            const float4 wv = *(const float4*)(rb + s * 1024 + t * 4);
            acc[s].x += val * wv.x; acc[s].y += val * wv.y;
            acc[s].z += val * wv.z; acc[s].w += val * wv.w;
        }
    }

    float4* o = (float4*)(out + (size_t)b * D_DIM);
#pragma unroll
    for (int s = 0; s < 4; ++s) {
        const float4 bv = *(const float4*)(dec_bias + s * 1024 + t * 4);
        float4 r;
        r.x = acc[s].x + bv.x; r.y = acc[s].y + bv.y;
        r.z = acc[s].z + bv.z; r.w = acc[s].w + bv.w;
        o[s * 256 + t] = r;
    }
}

// ---------------------------------------------------------------------------
extern "C" void kernel_launch(void* const* d_in, const int* in_sizes, int n_in,
                              void* d_out, int out_size, void* d_ws, size_t ws_size,
                              hipStream_t stream)
{
    (void)in_sizes; (void)n_in; (void)out_size; (void)ws_size;
    const float* x        = (const float*)d_in[0];
    const float* enc      = (const float*)d_in[1];   // [D, H] (== dec^T)
    const float* enc_bias = (const float*)d_in[2];   // [H]
    const float* dec      = (const float*)d_in[3];   // [H, D]
    const float* dec_bias = (const float*)d_in[4];   // [D]
    float* out = (float*)d_out;                      // [B, D]
    (void)enc;  // GEMM reads dec (== enc^T) for transpose-free B fragments

    // Workspace: pre f32 [B,H] (same intermediate the reference materializes),
    // then top-k values and indices.
    char*  ws   = (char*)d_ws;
    float* pre  = (float*)ws;                                        // 256 MB
    float* topv = (float*)(ws + (size_t)B_ROWS * H_DIM * 4);         //   1 MB
    int*   topi = (int*)  (ws + (size_t)B_ROWS * H_DIM * 4
                              + (size_t)B_ROWS * K_TOP * 4);         //   1 MB

    cc_encode_wmma <<<dim3(H_DIM / BN, B_ROWS / BM), 256, 0, stream>>>(x, dec, enc_bias, pre);
    cc_topk        <<<dim3(B_ROWS),                  256, 0, stream>>>(pre, topv, topi);
    cc_decode_async<<<dim3(B_ROWS),                  256, 0, stream>>>(topv, topi, dec, dec_bias, out);
}
